// CoolChicEncoder_77051713290353
// MI455X (gfx1250) — matrix operand order repaired and checked by reading, not compiled, not run on hardware
//
#include <hip/hip_runtime.h>
#include <math.h>

#define HH 1080
#define WWID 1920
#define HW (HH * WWID)

typedef __attribute__((ext_vector_type(16))) _Float16 v16h;
typedef __attribute__((ext_vector_type(8)))  _Float16 v8h;
typedef __attribute__((ext_vector_type(8)))  float    v8f;

// ---------------------------------------------------------------------------
// WMMA helper (gfx1250: V_WMMA_F32_16X16X32_F16, wave32)
// ---------------------------------------------------------------------------
__device__ inline v8f wmma16(v16h a, v16h b, v8f c) {
  return __builtin_amdgcn_wmma_f32_16x16x32_f16(false, a, false, b, (short)0, c,
                                                false, false);
}

// B fragment (K x N = 32x16 tile) from LDS float weights, row-major [din][dout]
__device__ inline v16h make_b(const float* Wm, int din, int dout, int ntile,
                              int lane) {
  int n  = (lane & 15) + (ntile << 4);
  int kb = (lane < 16) ? 0 : 8;
  v16h b;
#pragma unroll
  for (int h = 0; h < 16; ++h) {
    int k = kb + ((h < 8) ? h : h + 8);
    float v = (k < din && n < dout) ? Wm[k * dout + n] : 0.0f;
    b[h] = (_Float16)v;
  }
  return b;
}

// A fragment from a 16x16 f16 act tile (K 16..31 zero): one ds_load_b128/lane
__device__ inline v16h make_a16(const _Float16* act, int lane) {
  int m   = lane & 15;
  int off = (lane < 16) ? 0 : 8;
  v8h lo  = *(const v8h*)(act + m * 16 + off);
  v16h a;
#pragma unroll
  for (int h = 0; h < 8; ++h) {
    a[h]     = lo[h];
    a[h + 8] = (_Float16)0.0f;
  }
  return a;
}

// A fragment from a 16x32 f16 act tile: two ds_load_b128 per lane
__device__ inline v16h make_a32(const _Float16* act, int lane) {
  int m   = lane & 15;
  int off = (lane < 16) ? 0 : 8;
  v8h lo  = *(const v8h*)(act + m * 32 + off);
  v8h hi  = *(const v8h*)(act + m * 32 + 16 + off);
  v16h a;
#pragma unroll
  for (int h = 0; h < 8; ++h) {
    a[h]     = lo[h];
    a[h + 8] = hi[h];
  }
  return a;
}

// A fragment from a 16x8 f16 input tile (7 channels, slot 7 pre-zeroed)
__device__ inline v16h make_a_in(const _Float16* in, int lane) {
  int  m   = lane & 15;
  v8h  lo  = *(const v8h*)(in + m * 8);
  bool sel = lane < 16;
  v16h a;
#pragma unroll
  for (int h = 0; h < 8; ++h) {
    a[h]     = sel ? lo[h] : (_Float16)0.0f;
    a[h + 8] = (_Float16)0.0f;
  }
  return a;
}

// Store D (16x16 f32 frag) + bias (+ReLU) to LDS act tile as f16
__device__ inline void store_act(_Float16* act, int stride, int dout,
                                 const v8f& d, const float* bias, bool relu,
                                 int ncol_off, int lane) {
  int n  = (lane & 15) + ncol_off;
  int mo = (lane < 16) ? 0 : 8;
  float bv = (n < dout) ? bias[n] : 0.0f;
#pragma unroll
  for (int v = 0; v < 8; ++v) {
    float x = d[v] + bv;
    if (relu) x = fmaxf(x, 0.0f);
    act[(v + mo) * stride + n] = (_Float16)x;
  }
}

// ---------------------------------------------------------------------------
// Quantize: sent = round(latent * 2^gain)   (RNE, matches jnp.round)
// ---------------------------------------------------------------------------
__global__ __launch_bounds__(256) void quant_kernel(const float* __restrict__ in,
                                                    float* __restrict__ outp,
                                                    const float* __restrict__ lg,
                                                    int gi, int n) {
  float gain = exp2f(lg[gi]);
  for (int i = blockIdx.x * blockDim.x + threadIdx.x; i < n;
       i += gridDim.x * blockDim.x)
    outp[i] = rintf(in[i] * gain);
}

__global__ void init_rate(float* r) { *r = 0.0f; }

// ---------------------------------------------------------------------------
// ARM: 2D launch (blockIdx.y = row, 8 waves x 32-px x-tiles per block).
// Context gather -> 24-16-16-16-16-2 MLP via WMMA -> Laplace rate.
// Tap geometry precomputed per lane; no integer division in the hot path.
// ---------------------------------------------------------------------------
__global__ __launch_bounds__(256) void arm_kernel(
    const float* __restrict__ sent, int h, int w, int wtiles,
    const float* __restrict__ w0, const float* __restrict__ b0,
    const float* __restrict__ w1, const float* __restrict__ b1,
    const float* __restrict__ w2, const float* __restrict__ b2,
    const float* __restrict__ w3, const float* __restrict__ b3,
    const float* __restrict__ w4, const float* __restrict__ b4,
    float* __restrict__ rate_out) {
  __shared__ float sW[1184];  // w0:384 w1:256 w2:256 w3:256 w4:32
  __shared__ float sB[80];    // 16*4 + 16 (pad for b4)
  __shared__ __align__(16) _Float16 sAct[8][2][256];
  __shared__ float sMu[8][32];
  __shared__ float sLs[8][32];

  const int tid = threadIdx.x;
  for (int i = tid; i < 384; i += 256) sW[i] = w0[i];
  if (tid < 256) {
    sW[384 + tid] = w1[tid];
    sW[640 + tid] = w2[tid];
    sW[896 + tid] = w3[tid];
  }
  if (tid < 32) sW[1152 + tid] = w4[tid];
  if (tid < 16) {
    sB[tid]      = b0[tid];
    sB[16 + tid] = b1[tid];
    sB[32 + tid] = b2[tid];
    sB[48 + tid] = b3[tid];
    sB[64 + tid] = (tid < 2) ? b4[tid] : 0.0f;
  }
  __syncthreads();

  const int lane = tid & 31;
  const int wv   = tid >> 5;
  const int xt   = blockIdx.x * 8 + wv;

  float acc = 0.0f;
  if (xt < wtiles) {  // wave-uniform guard: EXEC stays full inside
    // Loop-invariant B fragments (K zero-padded to 32)
    v16h B0 = make_b(sW + 0,    24, 16, 0, lane);
    v16h B1 = make_b(sW + 384,  16, 16, 0, lane);
    v16h B2 = make_b(sW + 640,  16, 16, 0, lane);
    v16h B3 = make_b(sW + 896,  16, 16, 0, lane);
    v16h B4 = make_b(sW + 1152, 16,  2, 0, lane);

    const int y     = blockIdx.y;
    const int xbase = xt << 5;
    const int kb    = (lane < 16) ? 0 : 8;
    const int rowb  = y * w;

    // Per-lane tap geometry (computed once; k>=24 pad taps forced OOB via dx=w)
    int dyv[16], dxv[16], dof[16];
#pragma unroll
    for (int hh = 0; hh < 16; ++hh) {
      int k  = kb + ((hh < 8) ? hh : hh + 8);
      int dy = 0, dx = w;
      if (k < 24) {
        dy = k / 7 - 3;
        dx = k % 7 - 3;
      }
      dyv[hh] = dy;
      dxv[hh] = dx;
      dof[hh] = dy * w + dx;
    }

    // Prefetch next row of the latent (global_prefetch_b8)
    {
      int pfy = (y + 1 < h) ? (y + 1) : y;
      __builtin_prefetch(sent + pfy * w + ((xbase + lane < w) ? xbase + lane : 0),
                         0, 0);
    }

    const int m  = lane & 15;
    const int xA = xbase + m;
    const int xB = xA + 16;

    // Context fragments: unconditional clamped loads + select-to-zero
    v16h aA, aB;
#pragma unroll
    for (int hh = 0; hh < 16; ++hh) {
      int  cy  = y + dyv[hh];
      int  cxA = xA + dxv[hh];
      int  cxB = xB + dxv[hh];
      bool iA  = ((unsigned)cy < (unsigned)h) && ((unsigned)cxA < (unsigned)w);
      bool iB  = ((unsigned)cy < (unsigned)h) && ((unsigned)cxB < (unsigned)w);
      int  adA = iA ? (rowb + xA + dof[hh]) : 0;
      int  adB = iB ? (rowb + xB + dof[hh]) : 0;
      float vA = sent[adA];
      float vB = sent[adB];
      aA[hh] = (_Float16)(iA ? vA : 0.0f);
      aB[hh] = (_Float16)(iB ? vB : 0.0f);
    }

    v8f z  = {};
    v8f dA = wmma16(aA, B0, z);
    v8f dB = wmma16(aB, B0, z);
    store_act(&sAct[wv][0][0], 16, 16, dA, sB + 0, true, 0, lane);
    store_act(&sAct[wv][1][0], 16, 16, dB, sB + 0, true, 0, lane);

    aA = make_a16(&sAct[wv][0][0], lane);
    aB = make_a16(&sAct[wv][1][0], lane);
    dA = wmma16(aA, B1, z);
    dB = wmma16(aB, B1, z);
    store_act(&sAct[wv][0][0], 16, 16, dA, sB + 16, true, 0, lane);
    store_act(&sAct[wv][1][0], 16, 16, dB, sB + 16, true, 0, lane);

    aA = make_a16(&sAct[wv][0][0], lane);
    aB = make_a16(&sAct[wv][1][0], lane);
    dA = wmma16(aA, B2, z);
    dB = wmma16(aB, B2, z);
    store_act(&sAct[wv][0][0], 16, 16, dA, sB + 32, true, 0, lane);
    store_act(&sAct[wv][1][0], 16, 16, dB, sB + 32, true, 0, lane);

    aA = make_a16(&sAct[wv][0][0], lane);
    aB = make_a16(&sAct[wv][1][0], lane);
    dA = wmma16(aA, B3, z);
    dB = wmma16(aB, B3, z);
    store_act(&sAct[wv][0][0], 16, 16, dA, sB + 48, true, 0, lane);
    store_act(&sAct[wv][1][0], 16, 16, dB, sB + 48, true, 0, lane);

    aA = make_a16(&sAct[wv][0][0], lane);
    aB = make_a16(&sAct[wv][1][0], lane);
    dA = wmma16(aA, B4, z);
    dB = wmma16(aB, B4, z);

    // Extract mu (col 0) / log_scale (col 1) for both tiles
    {
      int n  = lane & 15;
      int mo = (lane < 16) ? 0 : 8;
      if (n < 2) {
        float bv = sB[64 + n];
#pragma unroll
        for (int v = 0; v < 8; ++v) {
          float va = dA[v] + bv;
          float vb = dB[v] + bv;
          if (n == 0) {
            sMu[wv][v + mo]      = va;
            sMu[wv][16 + v + mo] = vb;
          } else {
            sLs[wv][v + mo]      = va;
            sLs[wv][16 + v + mo] = vb;
          }
        }
      }
    }

    // Laplace rate: all 32 lanes, one pixel each
    {
      int px = xbase + lane;
      if (px < w) {
        float mu = sMu[wv][lane];
        float ls = fminf(fmaxf(sLs[wv][lane], -13.8155f), 13.8155f);
        float bs = expf(-0.5f * ls);
        float xv = sent[rowb + px];
        float d1 = xv + 0.5f - mu;
        float d0 = xv - 0.5f - mu;
        float s1 = (d1 >= 0.0f) ? 1.0f : -1.0f;
        float s0 = (d0 >= 0.0f) ? 1.0f : -1.0f;
        float c1 = 0.5f - 0.5f * s1 * expm1f(-fabsf(d1) / bs);
        float c0 = 0.5f - 0.5f * s0 * expm1f(-fabsf(d0) / bs);
        float p  = fmaxf(c1 - c0, 1.52587890625e-5f);  // 2^-16
        acc -= log2f(p);
      }
    }
  }

  // wave32 reduction + one atomic per active wave
#pragma unroll
  for (int o = 16; o > 0; o >>= 1) acc += __shfl_xor(acc, o, 32);
  if (lane == 0 && xt < wtiles) atomicAdd(rate_out, acc);
}

// ---------------------------------------------------------------------------
// SYN: 2D launch (blockIdx.y = row, 15 blocks x 8 waves = 120 x-tiles of 16).
// Bicubic upsample of 7 grids -> 7-32-32-32-3 MLP via WMMA -> RGB.
// ---------------------------------------------------------------------------
struct Grids {
  const float* g[7];
  int h[7];
  int w[7];
};

__device__ inline float cubw(float t) {
  const float A  = -0.5f;
  float       at = fabsf(t);
  if (at <= 1.0f) return ((A + 2.0f) * at - (A + 3.0f)) * at * at + 1.0f;
  if (at < 2.0f)  return (((at - 5.0f) * at + 8.0f) * at - 4.0f) * A;
  return 0.0f;
}

__device__ inline float bicubic(const float* g, int gh, int gw, int oy, int ox) {
  float sy = (oy + 0.5f) * ((float)gh / (float)HH)   - 0.5f;
  float sx = (ox + 0.5f) * ((float)gw / (float)WWID) - 0.5f;
  int   iy = (int)floorf(sy);
  int   ix = (int)floorf(sx);
  float fy = sy - (float)iy;
  float fx = sx - (float)ix;
  float wy[4], wx[4];
#pragma unroll
  for (int j = 0; j < 4; ++j) {
    wy[j] = cubw(fy + 1.0f - (float)j);
    wx[j] = cubw(fx + 1.0f - (float)j);
  }
  float acc = 0.0f;
#pragma unroll
  for (int j = 0; j < 4; ++j) {
    int   cy   = min(max(iy - 1 + j, 0), gh - 1);
    float rowa = 0.0f;
#pragma unroll
    for (int k = 0; k < 4; ++k) {
      int cx = min(max(ix - 1 + k, 0), gw - 1);
      rowa += wx[k] * g[cy * gw + cx];
    }
    acc += wy[j] * rowa;
  }
  return acc;
}

__global__ __launch_bounds__(256) void syn_kernel(
    Grids G,
    const float* __restrict__ w0, const float* __restrict__ b0,
    const float* __restrict__ w1, const float* __restrict__ b1,
    const float* __restrict__ w2, const float* __restrict__ b2,
    const float* __restrict__ w3, const float* __restrict__ b3,
    float* __restrict__ out) {
  __shared__ float sW[2368];  // w0:224 w1:1024 w2:1024 w3:96
  __shared__ float sB[128];   // 32*3 + 32 (pad for b3)
  __shared__ __align__(16) _Float16 sIn[8][16 * 8];
  __shared__ __align__(16) _Float16 sAct[8][16 * 32];

  const int tid = threadIdx.x;
  for (int i = tid; i < 224; i += 256)  sW[i]        = w0[i];
  for (int i = tid; i < 1024; i += 256) sW[224 + i]  = w1[i];
  for (int i = tid; i < 1024; i += 256) sW[1248 + i] = w2[i];
  if (tid < 96) sW[2272 + tid] = w3[tid];
  if (tid < 32) {
    sB[tid]      = b0[tid];
    sB[32 + tid] = b1[tid];
    sB[64 + tid] = b2[tid];
    sB[96 + tid] = (tid < 3) ? b3[tid] : 0.0f;
  }
  __syncthreads();

  const int lane = tid & 31;
  const int wv   = tid >> 5;

  // channel-7 slot stays zero for the vectorized A-frag load
  if (lane < 16) sIn[wv][lane * 8 + 7] = (_Float16)0.0f;

  v16h B0a = make_b(sW + 0,    7,  32, 0, lane);
  v16h B0b = make_b(sW + 0,    7,  32, 1, lane);
  v16h B1a = make_b(sW + 224,  32, 32, 0, lane);
  v16h B1b = make_b(sW + 224,  32, 32, 1, lane);
  v16h B2a = make_b(sW + 1248, 32, 32, 0, lane);
  v16h B2b = make_b(sW + 1248, 32, 32, 1, lane);
  v16h B3  = make_b(sW + 2272, 32,  3, 0, lane);

  const int y     = blockIdx.y;
  const int xt    = blockIdx.x * 8 + wv;  // 0..119, exact fit
  const int xbase = xt << 4;
  const int base  = y * WWID + xbase;

  // Cooperative gather: 16 pixels x 7 channels of bicubic samples
  for (int it = lane; it < 112; it += 32) {
    int c  = it >> 4;
    int mm = it & 15;
    sIn[wv][mm * 8 + c] =
        (_Float16)bicubic(G.g[c], G.h[c], G.w[c], y, xbase + mm);
  }

  v16h a  = make_a_in(&sIn[wv][0], lane);
  v8f  z  = {};
  v8f  d0 = wmma16(a, B0a, z);
  v8f  d1 = wmma16(a, B0b, z);
  store_act(&sAct[wv][0], 32, 32, d0, sB + 0, true, 0, lane);
  store_act(&sAct[wv][0], 32, 32, d1, sB + 0, true, 16, lane);

  a  = make_a32(&sAct[wv][0], lane);
  d0 = wmma16(a, B1a, z);
  d1 = wmma16(a, B1b, z);
  store_act(&sAct[wv][0], 32, 32, d0, sB + 32, true, 0, lane);
  store_act(&sAct[wv][0], 32, 32, d1, sB + 32, true, 16, lane);

  a  = make_a32(&sAct[wv][0], lane);
  d0 = wmma16(a, B2a, z);
  d1 = wmma16(a, B2b, z);
  store_act(&sAct[wv][0], 32, 32, d0, sB + 64, true, 0, lane);
  store_act(&sAct[wv][0], 32, 32, d1, sB + 64, true, 16, lane);

  a  = make_a32(&sAct[wv][0], lane);
  d0 = wmma16(a, B3, z);

  // Write clipped RGB (planar CHW)
  int n  = lane & 15;
  int mo = (lane < 16) ? 0 : 8;
  if (n < 3) {
    float bv = sB[96 + n];
#pragma unroll
    for (int v = 0; v < 8; ++v) {
      float val = fminf(fmaxf(d0[v] + bv, 0.0f), 1.0f);
      out[n * HW + base + v + mo] = val;
    }
  }
}

// ---------------------------------------------------------------------------
// Host launch
// ---------------------------------------------------------------------------
extern "C" void kernel_launch(void* const* d_in, const int* in_sizes, int n_in,
                              void* d_out, int out_size, void* d_ws,
                              size_t ws_size, hipStream_t stream) {
  (void)in_sizes; (void)n_in; (void)out_size; (void)ws_size;

  static const int gh[7] = {1080, 540, 270, 135, 68, 34, 17};
  static const int gw[7] = {1920, 960, 480, 240, 120, 60, 30};

  const float* latents[7];
  for (int i = 0; i < 7; ++i) latents[i] = (const float*)d_in[i];
  const float* lg = (const float*)d_in[7];
  const float *aw[5], *ab[5];
  for (int j = 0; j < 5; ++j) {
    aw[j] = (const float*)d_in[8 + 2 * j];
    ab[j] = (const float*)d_in[9 + 2 * j];
  }
  const float *sw[4], *sb[4];
  for (int j = 0; j < 4; ++j) {
    sw[j] = (const float*)d_in[18 + 2 * j];
    sb[j] = (const float*)d_in[19 + 2 * j];
  }

  float* out  = (float*)d_out;
  float* rate = out + (size_t)3 * HW;  // out_size = 3*H*W + 1

  // Workspace: quantized ("sent") grids, fp32, ~11.06 MB total
  float* ws = (float*)d_ws;
  float* sent[7];
  size_t off = 0;
  for (int i = 0; i < 7; ++i) {
    sent[i] = ws + off;
    off += (size_t)gh[i] * gw[i];
  }

  init_rate<<<1, 1, 0, stream>>>(rate);

  for (int i = 0; i < 7; ++i) {
    int n      = gh[i] * gw[i];
    int blocks = (n + 255) / 256;
    if (blocks > 8192) blocks = 8192;
    quant_kernel<<<blocks, 256, 0, stream>>>(latents[i], sent[i], lg, i, n);
  }

  for (int i = 0; i < 7; ++i) {
    int wtiles = (gw[i] + 31) / 32;
    dim3 grid((wtiles + 7) / 8, gh[i]);
    arm_kernel<<<grid, 256, 0, stream>>>(sent[i], gh[i], gw[i], wtiles,
                                         aw[0], ab[0], aw[1], ab[1],
                                         aw[2], ab[2], aw[3], ab[3],
                                         aw[4], ab[4], rate);
  }

  Grids G;
  for (int i = 0; i < 7; ++i) {
    G.g[i] = sent[i];
    G.h[i] = gh[i];
    G.w[i] = gw[i];
  }
  dim3 sgrid(15, HH);  // 15*8 waves = 120 x-tiles of 16 px = 1920
  syn_kernel<<<sgrid, 256, 0, stream>>>(G, sw[0], sb[0], sw[1], sb[1],
                                        sw[2], sb[2], sw[3], sb[3], out);
}
